// GenericMGU_91199335563543
// MI455X (gfx1250) — compile-verified, matrix operand order checked by reading, not run on hardware
//
#include <hip/hip_runtime.h>
#include <hip/hip_bf16.h>
#include <stdint.h>

// ---------------------------------------------------------------------------
// MGU (minimal gated unit), 3 layers, T=8192, IN=128, H=1024 on MI455X/gfx1250
//  - input projections: bf16 WMMA GEMM (128x128 tiles, f32 accumulate)
//  - recurrence: 16-workgroup persistent kernel, W_hh resident in LDS
//    (256KB/WGP via Tensor Data Mover), per-step matvec via v_wmma,
//    h exchanged through global double buffer + sense-reversing barrier
//    (plus s_cluster_barrier for a real cluster dispatch)
// ---------------------------------------------------------------------------

typedef __bf16 bf16_t;
typedef __attribute__((ext_vector_type(8)))  bf16_t       v8bf;
typedef __attribute__((ext_vector_type(16))) bf16_t       v16bf;
typedef __attribute__((ext_vector_type(8)))  float        v8f;
typedef __attribute__((ext_vector_type(4)))  unsigned int v4u;
typedef __attribute__((ext_vector_type(4)))  int          v4i;
typedef __attribute__((ext_vector_type(8)))  int          v8i;

#define T_STEPS 8192
#define IN_DIM  128
#define H_DIM   1024
#define G_DIM   2048      // 2*H
#define NBLK    16        // recurrence: one workgroup-cluster worth of WGPs
#define ROWS_PER_BLK 64   // h elements produced per block per step

// ------------------------------ small helpers ------------------------------

__device__ __forceinline__ v8f v8f_zero() {
  v8f z;
#pragma unroll
  for (int i = 0; i < 8; ++i) z[i] = 0.0f;
  return z;
}

__device__ __forceinline__ v16bf v16bf_zero() {
  v16bf z;
#pragma unroll
  for (int i = 0; i < 16; ++i) z[i] = (bf16_t)0.0f;
  return z;
}

// 16-bit A/B WMMA fragment (16x32 / 32x16) loader, row-major source.
// Per ISA layout: lanes 0-15 take K 0..7 & 16..23, lanes 16-31 take 8..15 & 24..31.
__device__ __forceinline__ v16bf load_frag(const bf16_t* rowp, int hs) {
  v8bf a0 = *(const v8bf*)(rowp + hs * 8);
  v8bf a1 = *(const v8bf*)(rowp + 16 + hs * 8);
  return __builtin_shufflevector(a0, a1, 0, 1, 2, 3, 4, 5, 6, 7,
                                 8, 9, 10, 11, 12, 13, 14, 15);
}

// 2D tile Global->LDS copy via the Tensor Data Mover (bf16 elements).
// D# built per CDNA5 ISA ch.8 (group0: count/lds/global/type, group1: dims).
__device__ __forceinline__ void tdm_load_tile_2d(unsigned lds_off,
                                                 const void* gaddr,
                                                 unsigned tile_rows,
                                                 unsigned tile_cols,
                                                 unsigned row_stride_elems) {
  const unsigned long long ga = (unsigned long long)(uintptr_t)gaddr;
  v4u g0;
  g0[0] = 1u;                                            // count=1, user mode
  g0[1] = lds_off;                                       // LDS byte address
  g0[2] = (unsigned)(ga & 0xFFFFFFFFull);                // global addr lo
  g0[3] = (unsigned)((ga >> 32) & 0x1FFFFFFull) | (2u << 30);  // addr hi | type=2
  v8i g1;
  g1[0] = (int)(1u << 16);                               // data_size=1 (2 bytes)
  g1[1] = (int)((tile_cols & 0xFFFFu) << 16);            // tensor_dim0 lo16
  g1[2] = (int)(((tile_cols >> 16) & 0xFFFFu) |
                ((tile_rows & 0xFFFFu) << 16));          // td0 hi | td1 lo
  g1[3] = (int)(((tile_rows >> 16) & 0xFFFFu) |
                ((tile_cols & 0xFFFFu) << 16));          // td1 hi | tile_dim0
  g1[4] = (int)(tile_rows & 0xFFFFu);                    // tile_dim1, tile_dim2=0
  g1[5] = (int)row_stride_elems;                         // tensor_dim0_stride lo32
  g1[6] = 0;
  g1[7] = 0;
  v4i g2; g2[0] = 0; g2[1] = 0; g2[2] = 0; g2[3] = 0;
  v4i g3; g3[0] = 0; g3[1] = 0; g3[2] = 0; g3[3] = 0;
#if defined(__clang_major__) && (__clang_major__ >= 23)
  v8i g4;
#pragma unroll
  for (int i = 0; i < 8; ++i) g4[i] = 0;
  __builtin_amdgcn_tensor_load_to_lds(g0, g1, g2, g3, g4, 0);
#else
  __builtin_amdgcn_tensor_load_to_lds(g0, g1, g2, g3, 0);
#endif
}

// ------------------------------ kernels ------------------------------------

__global__ void cvt_f32_bf16(const float* __restrict__ in,
                             bf16_t* __restrict__ out, long long n) {
  long long i = (long long)blockIdx.x * blockDim.x + threadIdx.x;
  const long long stride = (long long)gridDim.x * blockDim.x;
  for (; i < n; i += stride) out[i] = (bf16_t)in[i];
}

__global__ void bar_reset_kernel(unsigned* p) { p[0] = 0u; p[1] = 0u; }

// gi[T, 2H] = X[T, K](bf16) @ W[2H, K](bf16)^T + bias ; f32 out.
// Block: 256 threads (8 waves), 128x128 output tile, K chunked by 32.
__global__ void __launch_bounds__(256) mgu_input_proj(
    const bf16_t* __restrict__ X, const bf16_t* __restrict__ W,
    const float* __restrict__ bias, float* __restrict__ gi, int K) {
  __shared__ __align__(16) bf16_t Xs[128 * 32];
  __shared__ __align__(16) bf16_t Ws[128 * 32];

  const int tid = threadIdx.x;
  const int rowBase = blockIdx.y * 128;
  const int colBase = blockIdx.x * 128;
  const int wid = tid >> 5;
  const int lane = tid & 31;
  const int lr = lane & 15;
  const int hs = lane >> 4;
  const int wm = wid & 1;   // 0..1 -> 64-row half
  const int wn = wid >> 1;  // 0..3 -> 32-col quarter

  v8f acc[4][2];
#pragma unroll
  for (int mi = 0; mi < 4; ++mi)
#pragma unroll
    for (int ni = 0; ni < 2; ++ni) acc[mi][ni] = v8f_zero();

  const int nk = K >> 5;
  for (int kc = 0; kc < nk; ++kc) {
    __syncthreads();
#pragma unroll
    for (int s = tid; s < 512; s += 256) {
      const int r = s >> 2, c8 = s & 3;
      ((v8bf*)Xs)[s] =
          *(const v8bf*)(X + (size_t)(rowBase + r) * K + kc * 32 + c8 * 8);
      ((v8bf*)Ws)[s] =
          *(const v8bf*)(W + (size_t)(colBase + r) * K + kc * 32 + c8 * 8);
    }
    __syncthreads();

    v16bf bfrag[2];
#pragma unroll
    for (int ni = 0; ni < 2; ++ni)
      bfrag[ni] = load_frag(Ws + (wn * 32 + ni * 16 + lr) * 32, hs);
#pragma unroll
    for (int mi = 0; mi < 4; ++mi) {
      v16bf afrag = load_frag(Xs + (wm * 64 + mi * 16 + lr) * 32, hs);
#pragma unroll
      for (int ni = 0; ni < 2; ++ni)
        acc[mi][ni] = __builtin_amdgcn_wmma_f32_16x16x32_bf16(
            false, afrag, false, bfrag[ni], (short)0, acc[mi][ni], false, false);
    }
  }

#pragma unroll
  for (int mi = 0; mi < 4; ++mi)
#pragma unroll
    for (int ni = 0; ni < 2; ++ni) {
      const int col = colBase + wn * 32 + ni * 16 + lr;
      const float bv = bias[col];
#pragma unroll
      for (int j = 0; j < 8; ++j) {
        const int row = rowBase + wm * 64 + mi * 16 + hs * 8 + j;
        gi[(size_t)row * G_DIM + col] = acc[mi][ni][j] + bv;
      }
    }
}

// Persistent recurrence: grid = 16 blocks x 256 threads. Block b owns gate
// rows [64b,64b+64) (forget) and [1024+64b, ...) (candidate); W_hh slices
// live in LDS (2 x 128KB), h is exchanged via global double buffer.
__global__ void __launch_bounds__(256) mgu_recurrence(
    const float* __restrict__ gi, const bf16_t* __restrict__ whh16,
    const float* __restrict__ b_hh, const float* __restrict__ h0,
    float* __restrict__ hbuf, unsigned* __restrict__ bar,
    bf16_t* __restrict__ x_out) {
  extern __shared__ __align__(16) char smem[];
  bf16_t* Wf = (bf16_t*)smem;                   // 64 x 1024 bf16 = 128KB
  bf16_t* Wn = (bf16_t*)(smem + 131072);        // 64 x 1024 bf16 = 128KB
  float* yf = (float*)(smem + 262144);          // 64 f32
  float* yn = (float*)(smem + 262400);          // 64 f32
  bf16_t* hl = (bf16_t*)(smem + 262656);        // 1024 bf16

  const int blk = blockIdx.x;
  const int tid = threadIdx.x;
  const int wid = tid >> 5;
  const int lane = tid & 31;
  const int lr = lane & 15;
  const int hs = lane >> 4;

  // Stage both weight tiles with the Tensor Data Mover (wave 0 issues).
  if (tid < 32) {
    tdm_load_tile_2d(0u, whh16 + (size_t)(blk * ROWS_PER_BLK) * H_DIM,
                     ROWS_PER_BLK, H_DIM, H_DIM);
    tdm_load_tile_2d(131072u,
                     whh16 + (size_t)(H_DIM + blk * ROWS_PER_BLK) * H_DIM,
                     ROWS_PER_BLK, H_DIM, H_DIM);
    __builtin_amdgcn_s_wait_tensorcnt(0);
  }

  for (int i = tid; i < H_DIM; i += 256) hl[i] = (bf16_t)h0[i];
  float hprev = (tid < ROWS_PER_BLK) ? h0[blk * ROWS_PER_BLK + tid] : 0.0f;
  __syncthreads();

  const bf16_t* Wpart = (wid < 4) ? Wf : Wn;
  const int rb = wid & 3;  // 16-row block within the 64-row slice
  const bf16_t* arow_base = Wpart + (size_t)(rb * 16 + lr) * H_DIM;
  float* ypart = (wid < 4) ? yf : yn;

  volatile unsigned* bar_sense = bar + 1;

  for (int t = 0; t < T_STEPS; ++t) {
    // ---- gh slice via WMMA matvec: y[16 rows] = Wtile(16x1024) @ h ----
    v8f acc = v8f_zero();
#pragma unroll 4
    for (int kc = 0; kc < 32; ++kc) {
      v16bf a = load_frag(arow_base + kc * 32, hs);
      v16bf b = v16bf_zero();
      if (lr == 0) {  // h chunk occupies column 0 of B (lanes 0 and 16)
        const bf16_t* hp = hl + kc * 32 + hs * 8;
#pragma unroll
        for (int j = 0; j < 8; ++j) b[j] = hp[j];
#pragma unroll
        for (int j = 0; j < 8; ++j) b[8 + j] = hp[16 + j];
      }
      acc = __builtin_amdgcn_wmma_f32_16x16x32_bf16(false, a, false, b,
                                                    (short)0, acc, false, false);
    }
    if (lr == 0) {  // column 0 of D: lane0 -> rows 0..7, lane16 -> rows 8..15
      const int rbase = rb * 16 + hs * 8;
#pragma unroll
      for (int j = 0; j < 8; ++j) ypart[rbase + j] = acc[j];
    }
    __syncthreads();

    // ---- gate update for this block's 64 h elements ----
    if (tid < ROWS_PER_BLK) {
      const int col = blk * ROWS_PER_BLK + tid;
      const float gf = gi[(size_t)t * G_DIM + col];
      const float gn = gi[(size_t)t * G_DIM + H_DIM + col];
      const float hf = yf[tid] + b_hh[col];
      const float hn = yn[tid] + b_hh[H_DIM + col];
      const float f = 1.0f / (1.0f + __expf(-(gf + hf)));
      const float n = tanhf(gn + f * hn);
      const float hy = (1.0f - f) * hprev + f * n;  // n + (1-f)(h-n)
      hprev = hy;
      hbuf[(size_t)(t & 1) * H_DIM + col] = hy;
      x_out[(size_t)t * H_DIM + col] = (bf16_t)hy;
      if (t + 1 < T_STEPS) {
        __builtin_prefetch(gi + (size_t)(t + 1) * G_DIM + col, 0, 1);
        __builtin_prefetch(gi + (size_t)(t + 1) * G_DIM + H_DIM + col, 0, 1);
      }
    }

    // ---- barrier across the 16 resident workgroups ----
    __threadfence();
    __syncthreads();
    __builtin_amdgcn_s_cluster_barrier();  // cluster dispatch: HW barrier; else NOP
    if (tid == 0) {
      const unsigned target = (unsigned)(t + 1);
      const unsigned prev = atomicAdd(bar, 1u);
      if (prev == NBLK - 1) {
        *bar = 0u;
        __threadfence();
        *bar_sense = target;
      } else {
        while (*bar_sense != target) __builtin_amdgcn_s_sleep(1);
      }
    }
    __syncthreads();

    // ---- pull the full new h into LDS as bf16 for the next step ----
    const volatile float* hv = hbuf + (size_t)(t & 1) * H_DIM;
    for (int i = tid; i < H_DIM; i += 256) hl[i] = (bf16_t)hv[i];
    __syncthreads();
  }
}

// out[t] = x[t,:] . w_out + b_out ; one wave per timestep.
__global__ void __launch_bounds__(256) mgu_out_proj(
    const bf16_t* __restrict__ x, const float* __restrict__ w_out,
    const float* __restrict__ b_out, float* __restrict__ out) {
  const int t = (int)((blockIdx.x * 256 + threadIdx.x) >> 5);
  const int lane = threadIdx.x & 31;
  if (t >= T_STEPS) return;
  const bf16_t* xr = x + (size_t)t * H_DIM;
  float s = 0.0f;
#pragma unroll 8
  for (int i = lane; i < H_DIM; i += 32) s += (float)xr[i] * w_out[i];
#pragma unroll
  for (int off = 16; off > 0; off >>= 1) s += __shfl_down(s, off, 32);
  if (lane == 0) out[t] = s + b_out[0];
}

// ------------------------------ host driver --------------------------------

static inline size_t align256(size_t x) { return (x + 255) & ~(size_t)255; }

extern "C" void kernel_launch(void* const* d_in, const int* in_sizes, int n_in,
                              void* d_out, int out_size, void* d_ws,
                              size_t ws_size, hipStream_t stream) {
  (void)in_sizes; (void)n_in; (void)out_size; (void)ws_size;

  const float* S     = (const float*)d_in[0];   // [T, IN]
  const float* h0    = (const float*)d_in[1];   // [L, H]
  const float* wih0  = (const float*)d_in[2];   // [2H, IN]
  const float* whh0  = (const float*)d_in[3];   // [2H, H]
  const float* bih0  = (const float*)d_in[4];   // [2H]
  const float* bhh0  = (const float*)d_in[5];   // [2H]
  const float* wihR  = (const float*)d_in[6];   // [2, 2H, H]
  const float* whhR  = (const float*)d_in[7];   // [2, 2H, H]
  const float* bihR  = (const float*)d_in[8];   // [2, 2H]
  const float* bhhR  = (const float*)d_in[9];   // [2, 2H]
  const float* wout  = (const float*)d_in[10];  // [1, H]
  const float* bout  = (const float*)d_in[11];  // [1]
  float* outp        = (float*)d_out;           // [T]

  char* ws = (char*)d_ws;
  size_t o = 0;
  const size_t o_S16  = o; o += align256((size_t)T_STEPS * IN_DIM * 2);
  const size_t o_wih0 = o; o += align256((size_t)G_DIM * IN_DIM * 2);
  const size_t o_wihR = o; o += align256((size_t)2 * G_DIM * H_DIM * 2);
  const size_t o_whh  = o; o += align256((size_t)3 * G_DIM * H_DIM * 2);
  const size_t o_gi   = o; o += align256((size_t)T_STEPS * G_DIM * 4);
  const size_t o_x0   = o; o += align256((size_t)T_STEPS * H_DIM * 2);
  const size_t o_x1   = o; o += align256((size_t)T_STEPS * H_DIM * 2);
  const size_t o_h    = o; o += align256((size_t)2 * H_DIM * 4);
  const size_t o_bar  = o; o += 256;

  bf16_t* S16   = (bf16_t*)(ws + o_S16);
  bf16_t* WIH0  = (bf16_t*)(ws + o_wih0);
  bf16_t* WIHR  = (bf16_t*)(ws + o_wihR);
  bf16_t* WHH   = (bf16_t*)(ws + o_whh);
  float*  GI    = (float*)(ws + o_gi);
  bf16_t* X0    = (bf16_t*)(ws + o_x0);
  bf16_t* X1    = (bf16_t*)(ws + o_x1);
  float*  HBUF  = (float*)(ws + o_h);
  unsigned* BAR = (unsigned*)(ws + o_bar);

  // bf16 conversions
  cvt_f32_bf16<<<1024, 256, 0, stream>>>(S, S16, (long long)T_STEPS * IN_DIM);
  cvt_f32_bf16<<<512, 256, 0, stream>>>(wih0, WIH0, (long long)G_DIM * IN_DIM);
  cvt_f32_bf16<<<2048, 256, 0, stream>>>(wihR, WIHR, (long long)2 * G_DIM * H_DIM);
  cvt_f32_bf16<<<1024, 256, 0, stream>>>(whh0, WHH, (long long)G_DIM * H_DIM);
  cvt_f32_bf16<<<2048, 256, 0, stream>>>(
      whhR, WHH + (size_t)G_DIM * H_DIM, (long long)2 * G_DIM * H_DIM);

  const dim3 projGrid(G_DIM / 128, T_STEPS / 128);  // (16, 64)
  const size_t REC_SMEM = 262144 + 256 + 256 + 2048;  // 264704 B

  // ---- layer 0 ----
  mgu_input_proj<<<projGrid, 256, 0, stream>>>(S16, WIH0, bih0, GI, IN_DIM);
  bar_reset_kernel<<<1, 1, 0, stream>>>(BAR);
  mgu_recurrence<<<NBLK, 256, REC_SMEM, stream>>>(
      GI, WHH, bhh0, h0, HBUF, BAR, X0);

  // ---- layer 1 ----
  mgu_input_proj<<<projGrid, 256, 0, stream>>>(X0, WIHR, bihR, GI, H_DIM);
  bar_reset_kernel<<<1, 1, 0, stream>>>(BAR);
  mgu_recurrence<<<NBLK, 256, REC_SMEM, stream>>>(
      GI, WHH + (size_t)G_DIM * H_DIM, bhhR, h0 + H_DIM, HBUF, BAR, X1);

  // ---- layer 2 ----
  mgu_input_proj<<<projGrid, 256, 0, stream>>>(
      X1, WIHR + (size_t)G_DIM * H_DIM, bihR + G_DIM, GI, H_DIM);
  bar_reset_kernel<<<1, 1, 0, stream>>>(BAR);
  mgu_recurrence<<<NBLK, 256, REC_SMEM, stream>>>(
      GI, WHH + (size_t)2 * G_DIM * H_DIM, bhhR + G_DIM, h0 + 2 * H_DIM,
      HBUF, BAR, X0);

  // ---- output projection ----
  mgu_out_proj<<<T_STEPS / 8, 256, 0, stream>>>(X0, wout, bout, outp);
}
